// EvoformerStack_29841432773275
// MI455X (gfx1250) — matrix operand order repaired and checked by reading, not compiled, not run on hardware
//
#include <hip/hip_runtime.h>

#define DEVINL __device__ __forceinline__

typedef __attribute__((ext_vector_type(8)))  float        v8f;
typedef __attribute__((ext_vector_type(8)))  unsigned int v8u;
typedef __attribute__((ext_vector_type(16))) __bf16       v16bf;

DEVINL unsigned short f2bf(float f) {
  unsigned int u = __float_as_uint(f);
  u += 0x7fffu + ((u >> 16) & 1u);   // round-to-nearest-even
  return (unsigned short)(u >> 16);
}
DEVINL float wsum(float v) {
#pragma unroll
  for (int m = 16; m; m >>= 1) v += __shfl_xor(v, m, 32);
  return v;
}
DEVINL float wmaxf(float v) {
#pragma unroll
  for (int m = 16; m; m >>= 1) v = fmaxf(v, __shfl_xor(v, m, 32));
  return v;
}

// ---------------------------------------------------------------------------
// Generic batched bf16 WMMA GEMM.
//   C[b][m][n] = act(alpha * sum_k A[b][m][k] * Bt[b][n][k] + bias[n]) (+=)
// A row-major bf16 (lda), Bt = B^T row-major bf16 (ldb).
// Batch offset = (b/inner)*Outer + (b%inner)*Inner for A, Bt, C separately.
// Block = 256 threads = 8 waves, 2(M) x 4(N); wave computes 32x32 via 2x2
// WMMA tiles (A frags reused across col-tiles, B frags across row-tiles).
// Block tile = 64 x 128.  M must be a multiple of 64 (true for all calls).
// Out-of-range N columns: load from a clamped address (their product columns
// are lane-private and never stored), so no divergent predication.
// Fragment layouts per CDNA5 ISA 7.12.2 (16-bit A 16x32, B 32x16).
// ---------------------------------------------------------------------------
__global__ __launch_bounds__(256)
void k_gemm(const unsigned short* __restrict__ A, long long lda, long long aO, long long aI,
            const unsigned short* __restrict__ Bt, long long ldb, long long bO, long long bI,
            const float* __restrict__ bias,
            float* __restrict__ Cf, unsigned short* __restrict__ Cb,
            long long ldc, long long cO, long long cI, int inner,
            int M, int N, int K, float alpha, int act, int accum) {
  (void)M;
  int b  = blockIdx.z;
  int bo = b / inner, bi = b - bo * inner;
  long long aOff = (long long)bo * aO + (long long)bi * aI;
  long long bOff = (long long)bo * bO + (long long)bi * bI;
  long long cOff = (long long)bo * cO + (long long)bi * cI;

  int tid  = threadIdx.x;
  int wid  = tid >> 5, lane = tid & 31;
  int m0   = blockIdx.y * 64 + (wid & 1) * 32;     // wave rows: m0..m0+31
  int n0   = blockIdx.x * 128 + (wid >> 1) * 32;   // wave cols: n0..n0+31
  int hf   = lane >> 4, l15 = lane & 15;

  const unsigned short* ap[2];
  const unsigned short* bp[2];
  int ncol[2];
#pragma unroll
  for (int i = 0; i < 2; i++) {
    int mrow = m0 + i * 16 + l15;
    ap[i] = A + aOff + (long long)mrow * lda;
    int nc = n0 + i * 16 + l15;
    ncol[i] = nc;
    int ncl = nc < N ? nc : (N - 1);               // clamp: safe garbage load
    bp[i] = Bt + bOff + (long long)ncl * ldb;
  }

  v8f acc[2][2];
#pragma unroll
  for (int i = 0; i < 2; i++)
#pragma unroll
    for (int j = 0; j < 2; j++) acc[i][j] = v8f{0.f,0.f,0.f,0.f,0.f,0.f,0.f,0.f};

  for (int kk = 0; kk < K; kk += 32) {
    v16bf af[2], bf[2];
#pragma unroll
    for (int i = 0; i < 2; i++) {
      v8u au, bu;
#pragma unroll
      for (int v = 0; v < 8; v++) {
        int ka = ((v & 4) << 2) + hf * 8 + ((v & 3) << 1);  // A K-pair offset
        au[v]  = *(const unsigned int*)(ap[i] + kk + ka);
      }
#pragma unroll
      for (int v = 0; v < 8; v++) {
        int kb = ((v & 4) << 2) + hf * 8 + ((v & 3) << 1);  // B K-pair offset
        bu[v]  = *(const unsigned int*)(bp[i] + kk + kb);
      }
      af[i] = __builtin_bit_cast(v16bf, au);
      bf[i] = __builtin_bit_cast(v16bf, bu);
    }
#pragma unroll
    for (int i = 0; i < 2; i++)
#pragma unroll
      for (int j = 0; j < 2; j++)
        acc[i][j] = __builtin_amdgcn_wmma_f32_16x16x32_bf16(
            false, af[i], false, bf[j], (short)0, acc[i][j], false, false);
  }

#pragma unroll
  for (int j = 0; j < 2; j++) {
    if (ncol[j] >= N) continue;
    float bv = bias ? bias[ncol[j]] : 0.f;
#pragma unroll
    for (int i = 0; i < 2; i++) {
#pragma unroll
      for (int r = 0; r < 8; r++) {
        int row = m0 + i * 16 + hf * 8 + r;                 // C/D layout 7.12.2
        float v = acc[i][j][r] * alpha + bv;
        if (act == 1)      v = fmaxf(v, 0.f);
        else if (act == 2) v = 1.f / (1.f + __expf(-v));
        long long idx = cOff + (long long)row * ldc + ncol[j];
        if (accum) v += Cf[idx];
        if (Cf) Cf[idx] = v;
        if (Cb) Cb[idx] = f2bf(v);
      }
    }
  }
}

// LayerNorm over last dim, one wave per row, rows multiple of 8. bf16 out.
__global__ __launch_bounds__(256)
void k_ln(const float* __restrict__ x, const float* __restrict__ g,
          const float* __restrict__ b, unsigned short* __restrict__ out, int C) {
  int wid = threadIdx.x >> 5, lane = threadIdx.x & 31;
  long long row = (long long)blockIdx.x * 8 + wid;
  const float* xr = x + row * C;
  float s = 0.f, s2 = 0.f;
  for (int c = lane; c < C; c += 32) { float v = xr[c]; s += v; s2 += v * v; }
  s = wsum(s); s2 = wsum(s2);
  float mu = s / C;
  float rs = rsqrtf(s2 / C - mu * mu + 1e-5f);
  unsigned short* o = out + row * C;
  for (int c = lane; c < C; c += 32) o[c] = f2bf((xr[c] - mu) * rs * g[c] + b[c]);
}

// LayerNorm where channel c lives at x[c*stride + r] (tri-mult einsum output).
__global__ __launch_bounds__(256)
void k_ln_strided(const float* __restrict__ x, long long stride,
                  const float* __restrict__ g, const float* __restrict__ b,
                  unsigned short* __restrict__ out, int C, long long R) {
  long long r = (long long)blockIdx.x * 256 + threadIdx.x;
  if (r >= R) return;
  float s = 0.f, s2 = 0.f;
  for (int c = 0; c < C; c++) { float v = x[c * stride + r]; s += v; s2 += v * v; }
  float mu = s / C;
  float rs = rsqrtf(s2 / C - mu * mu + 1e-5f);
  for (int c = 0; c < C; c++) {
    float v = x[c * stride + r];
    out[r * C + c] = f2bf((v - mu) * rs * g[c] + b[c]);
  }
}

// Softmax over keys (L=256 fixed), fusing mask bias + per-head pair bias.
// row = b*L + q, b = o*H + h.  bf16 probabilities out.
__global__ __launch_bounds__(256)
void k_softmax(const float* __restrict__ sc, unsigned short* __restrict__ pr,
               const float* __restrict__ mask2d, int mld, int maskT,
               const float* __restrict__ hb, int H) {
  const int L = 256;
  int wid = threadIdx.x >> 5, lane = threadIdx.x & 31;
  long long row = (long long)blockIdx.x * 8 + wid;
  long long b = row >> 8;
  int q = (int)(row & 255);
  int o = (int)(b / H), h = (int)(b % H);
  const float* sr = sc + row * L;
  float vals[8];
  float mx = -3.0e38f;
#pragma unroll
  for (int t = 0; t < 8; t++) {
    int k = lane + t * 32;
    float mk = maskT ? mask2d[(long long)k * mld + o] : mask2d[(long long)o * mld + k];
    float v = sr[k] + 5.0e4f * (mk - 1.f) + hb[((long long)q * L + k) * H + h];
    vals[t] = v;
    mx = fmaxf(mx, v);
  }
  mx = wmaxf(mx);
  float sum = 0.f;
#pragma unroll
  for (int t = 0; t < 8; t++) { vals[t] = __expf(vals[t] - mx); sum += vals[t]; }
  sum = wsum(sum);
  float inv = 1.f / sum;
  unsigned short* pw = pr + row * L;
#pragma unroll
  for (int t = 0; t < 8; t++) pw[lane + t * 32] = f2bf(vals[t] * inv);
}

// Weight transpose+convert: out[N,K] bf16 = w[K,N] f32.
__global__ void k_wT(const float* __restrict__ w, unsigned short* __restrict__ o,
                     int K, int N, long long total) {
  long long t = (long long)blockIdx.x * 256 + threadIdx.x;
  if (t >= total) return;
  long long n = t / K, k = t - n * K;
  o[t] = f2bf(w[k * N + n]);
}
__global__ void k_cvt(const float* __restrict__ a, unsigned short* __restrict__ o, long long n) {
  long long t = (long long)blockIdx.x * 256 + threadIdx.x;
  if (t < n) o[t] = f2bf(a[t]);
}
__global__ void k_mul_b(const float* __restrict__ a, const float* __restrict__ b,
                        unsigned short* __restrict__ o, long long n) {
  long long t = (long long)blockIdx.x * 256 + threadIdx.x;
  if (t < n) o[t] = f2bf(a[t] * b[t]);
}
__global__ void k_add_mul(float* __restrict__ d, const float* __restrict__ a,
                          const float* __restrict__ b, long long n) {
  long long t = (long long)blockIdx.x * 256 + threadIdx.x;
  if (t < n) d[t] += a[t] * b[t];
}
__global__ void k_add_masked(float* __restrict__ d, const float* __restrict__ s,
                             const float* __restrict__ m, int C, long long n) {
  long long t = (long long)blockIdx.x * 256 + threadIdx.x;
  if (t < n) d[t] += s[t] * m[t / C];
}
// v bf16 [(o*L+l), H*Ca] -> vT bf16 [((o*H+h)*Ca+c)*L + l]
__global__ void k_head_pack(const unsigned short* __restrict__ in,
                            unsigned short* __restrict__ o,
                            int L, int H, int Ca, long long total) {
  long long t = (long long)blockIdx.x * 256 + threadIdx.x;
  if (t >= total) return;
  int l = (int)(t % L);
  long long r = t / L;
  int c = (int)(r % Ca); r /= Ca;
  int h = (int)(r % H);
  long long ob = r / H;
  o[t] = in[(ob * L + l) * (long long)(H * Ca) + h * Ca + c];
}
// OPM: a f32 [(s*256+i),32] * mask[s,i] -> bf16 [(i*32+c)*64+s]
__global__ void k_opm_pack(const float* __restrict__ in, const float* __restrict__ mask,
                           unsigned short* __restrict__ o, long long total) {
  long long t = (long long)blockIdx.x * 256 + threadIdx.x;
  if (t >= total) return;
  int s = (int)(t & 63);
  long long r = t >> 6;
  int c = (int)(r & 31);
  long long i = r >> 5;
  long long ri = (long long)s * 256 + i;
  o[t] = f2bf(in[ri * 32 + c] * mask[ri]);
}
// outer f32 [(i*32+c), (j*32+d)] (ld 8192) -> bf16 [((i*256+j)*32+c)*32+d]
__global__ void k_opm_perm(const float* __restrict__ in, unsigned short* __restrict__ o,
                           long long total) {
  long long t = (long long)blockIdx.x * 256 + threadIdx.x;
  if (t >= total) return;
  int d = (int)(t & 31);
  long long r = t >> 5;
  int c = (int)(r & 31); r >>= 5;
  int j = (int)(r & 255);
  long long i = r >> 8;
  o[t] = f2bf(in[(i * 32 + c) * 8192 + (long long)j * 32 + d]);
}
__global__ void k_norm(const float* __restrict__ mask, float* __restrict__ nrm) {
  int t = blockIdx.x * 256 + threadIdx.x;
  int i = t >> 8, j = t & 255;
  float s = 0.f;
  for (int ss = 0; ss < 64; ss++) s += mask[ss * 256 + i] * mask[ss * 256 + j];
  nrm[t] = s;
}
__global__ void k_opm_add(float* __restrict__ z, const float* __restrict__ x,
                          const float* __restrict__ nrm, long long n) {
  long long t = (long long)blockIdx.x * 256 + threadIdx.x;
  if (t < n) z[t] += x[t] / (1e-3f + nrm[t >> 7]);
}
// tri-mult operand pack: out bf16 [(c*256+i)*256+k] = mask[r]*p[r,c]*gsig[r,c],
// r = trans ? k*256+i : i*256+k   (C=128, N=256 fixed)
__global__ void k_trimul_pack(const float* __restrict__ p, const float* __restrict__ g,
                              const float* __restrict__ mask, unsigned short* __restrict__ o,
                              int trans, long long total) {
  long long t = (long long)blockIdx.x * 256 + threadIdx.x;
  if (t >= total) return;
  int k = (int)(t & 255);
  long long r = t >> 8;
  int i = (int)(r & 255);
  long long c = r >> 8;
  long long rr = trans ? ((long long)k * 256 + i) : ((long long)i * 256 + k);
  o[t] = f2bf(mask[rr] * p[rr * 128 + c] * g[rr * 128 + c]);
}
// z [i,j,c] <-> [j,i,c], C=128, 256x256
__global__ void k_transpose3d(const float* __restrict__ in, float* __restrict__ o,
                              long long total) {
  long long t = (long long)blockIdx.x * 256 + threadIdx.x;
  if (t >= total) return;
  int c = (int)(t & 127);
  long long r = t >> 7;
  int j = (int)(r & 255);
  long long i = r >> 8;
  o[((long long)j * 256 + i) * 128 + c] = in[t];
}

// ---------------------------------------------------------------------------
// Host helpers
// ---------------------------------------------------------------------------
static void gemm(hipStream_t st,
                 const void* A, long long lda, long long aO, long long aI,
                 const void* Bt, long long ldb, long long bO, long long bI,
                 const float* bias, float* Cf, void* Cb,
                 long long ldc, long long cO, long long cI, int inner,
                 int M, int N, int K, int batches, float alpha, int act, int accum) {
  dim3 g((N + 127) / 128, M / 64, batches);
  k_gemm<<<g, 256, 0, st>>>((const unsigned short*)A, lda, aO, aI,
                            (const unsigned short*)Bt, ldb, bO, bI, bias,
                            Cf, (unsigned short*)Cb, ldc, cO, cI, inner,
                            M, N, K, alpha, act, accum);
}

// Triangle attention (shared by tas / tae). Rows = 65536, C_Z=128, H=4, c=32.
static void tri_att_run(hipStream_t st, const float* zsrc, float* zdst,
                        const float* pmask, int maskT,
                        const float* lnB, const float* lnG,
                        const unsigned short* WQ, const unsigned short* WK,
                        const unsigned short* WV, const unsigned short* WG,
                        const float* gB, const unsigned short* WO, const float* oB,
                        const unsigned short* WTB,
                        char* slabA, char* slabB, char* buf) {
  const int Rr = 65536, C = 128, L = 256, H = 4, Ca = 32;
  size_t cur = 0;
  auto al = [&](size_t bytes) { void* p = buf + cur; cur += (bytes + 255) & ~(size_t)255; return p; };
  unsigned short* lnb   = (unsigned short*)al(16777216);
  unsigned short* qb    = (unsigned short*)al(16777216);
  unsigned short* kb    = (unsigned short*)al(16777216);
  unsigned short* vb    = (unsigned short*)al(16777216);
  float*          tb    = (float*)al(1048576);
  float*          gate  = (float*)al(33554432);
  unsigned short* vT    = (unsigned short*)al(16777216);
  float*          att   = (float*)al(33554432);
  unsigned short* gated = (unsigned short*)al(16777216);
  float*          scores = (float*)slabA;
  unsigned short* probs  = (unsigned short*)slabB;

  k_ln<<<Rr / 8, 256, 0, st>>>(zsrc, lnG, lnB, lnb, C);
  gemm(st, lnb, C, 0, 0, WQ, C, 0, 0, nullptr, nullptr, qb, C, 0, 0, 1, Rr, C, C, 1, 1.f, 0, 0);
  gemm(st, lnb, C, 0, 0, WK, C, 0, 0, nullptr, nullptr, kb, C, 0, 0, 1, Rr, C, C, 1, 1.f, 0, 0);
  gemm(st, lnb, C, 0, 0, WV, C, 0, 0, nullptr, nullptr, vb, C, 0, 0, 1, Rr, C, C, 1, 1.f, 0, 0);
  gemm(st, lnb, C, 0, 0, WTB, C, 0, 0, nullptr, tb, nullptr, H, 0, 0, 1, Rr, H, C, 1, 1.f, 0, 0);
  gemm(st, lnb, C, 0, 0, WG, C, 0, 0, gB, gate, nullptr, C, 0, 0, 1, Rr, C, C, 1, 1.f, 2, 0);
  // scores[o*H+h][q][k] = (1/sqrt(32)) q . k
  gemm(st, qb, C, (long long)L * C, Ca, kb, C, (long long)L * C, Ca,
       nullptr, scores, nullptr, L, (long long)H * L * L, (long long)L * L, H,
       L, L, Ca, 256 * H, 0.17677669529663687f, 0, 0);
  k_softmax<<<(256 * H * L) / 8, 256, 0, st>>>(scores, probs, pmask, 256, maskT, tb, H);
  k_head_pack<<<32768, 256, 0, st>>>(vb, vT, L, H, Ca, 8388608LL);
  gemm(st, probs, L, (long long)H * L * L, (long long)L * L,
       vT, L, (long long)H * Ca * L, (long long)Ca * L,
       nullptr, att, nullptr, C, (long long)L * C, Ca, H,
       L, Ca, L, 256 * H, 1.f, 0, 0);
  k_mul_b<<<32768, 256, 0, st>>>(att, gate, gated, (long long)Rr * C);
  gemm(st, gated, C, 0, 0, WO, C, 0, 0, oB, zdst, nullptr, C, 0, 0, 1, Rr, C, C, 1, 1.f, 0, 1);
}

// Triangle multiplication (tmo: trans=0, tmi: trans=1).
static void tri_mul_run(hipStream_t st, float* z, const float* pmask,
                        const float* lniB, const float* lniG,
                        const unsigned short* WAP, const float* apB,
                        const unsigned short* WAG, const float* agB,
                        const unsigned short* WBP, const float* bpB,
                        const unsigned short* WBG, const float* bgB,
                        const unsigned short* WG, const float* gB,
                        const float* lnoB, const float* lnoG,
                        const unsigned short* WP, const float* pB,
                        int trans, char* slabA, char* slabB, char* buf) {
  const int Rr = 65536, C = 128, N = 256;
  size_t cur = 0;
  auto al = [&](size_t bytes) { void* p = buf + cur; cur += (bytes + 255) & ~(size_t)255; return p; };
  unsigned short* lnin = (unsigned short*)al(16777216);
  float* ap   = (float*)al(33554432);
  float* ag   = (float*)al(33554432);
  float* bp   = (float*)al(33554432);
  float* bg   = (float*)al(33554432);
  float* gate = (float*)al(33554432);
  unsigned short* aT  = (unsigned short*)al(16777216);
  unsigned short* bT  = (unsigned short*)al(16777216);
  unsigned short* xln = (unsigned short*)al(16777216);
  float* xbuf = (float*)slabB;   // [128][256][256] f32
  float* pout = (float*)slabA;

  k_ln<<<Rr / 8, 256, 0, st>>>(z, lniG, lniB, lnin, C);
  gemm(st, lnin, C, 0, 0, WAP, C, 0, 0, apB, ap, nullptr, C, 0, 0, 1, Rr, C, C, 1, 1.f, 0, 0);
  gemm(st, lnin, C, 0, 0, WAG, C, 0, 0, agB, ag, nullptr, C, 0, 0, 1, Rr, C, C, 1, 1.f, 2, 0);
  gemm(st, lnin, C, 0, 0, WBP, C, 0, 0, bpB, bp, nullptr, C, 0, 0, 1, Rr, C, C, 1, 1.f, 0, 0);
  gemm(st, lnin, C, 0, 0, WBG, C, 0, 0, bgB, bg, nullptr, C, 0, 0, 1, Rr, C, C, 1, 1.f, 2, 0);
  gemm(st, lnin, C, 0, 0, WG, C, 0, 0, gB, gate, nullptr, C, 0, 0, 1, Rr, C, C, 1, 1.f, 2, 0);
  long long total = (long long)C * N * N;   // 8,388,608
  k_trimul_pack<<<32768, 256, 0, st>>>(ap, ag, pmask, aT, trans, total);
  k_trimul_pack<<<32768, 256, 0, st>>>(bp, bg, pmask, bT, trans, total);
  // per-channel einsum: x[c][i][j] = sum_k aT[c][i][k]*bT[c][j][k]
  gemm(st, aT, N, (long long)N * N, 0, bT, N, (long long)N * N, 0,
       nullptr, xbuf, nullptr, N, (long long)N * N, 0, 1,
       N, N, N, C, 1.f, 0, 0);
  k_ln_strided<<<Rr / 256, 256, 0, st>>>(xbuf, (long long)Rr, lnoG, lnoB, xln, C, (long long)Rr);
  gemm(st, xln, C, 0, 0, WP, C, 0, 0, pB, pout, nullptr, C, 0, 0, 1, Rr, C, C, 1, 1.f, 0, 0);
  k_add_mul<<<32768, 256, 0, st>>>(z, pout, gate, (long long)Rr * C);
}

// ---------------------------------------------------------------------------
// Parameter leaf indices (jax tree-flatten order: dict keys alphabetical).
// d_in: 0=m, 1=z, 2=msa_mask, 3=pair_mask, 4..89 = params leaves.
// ---------------------------------------------------------------------------
extern "C" void kernel_launch(void* const* d_in, const int* in_sizes, int n_in,
                              void* d_out, int out_size, void* d_ws, size_t ws_size,
                              hipStream_t stream) {
  (void)in_sizes; (void)n_in; (void)out_size; (void)ws_size;
  const int S = 64, R = 256, CM = 256, CZ = 128;
  auto f = [&](int i) { return (const float*)d_in[i]; };

  // outputs live in d_out: m | z | s
  float* Mo = (float*)d_out;
  float* Zo = Mo + (size_t)S * R * CM;       // 4,194,304 floats
  float* So = Zo + (size_t)R * R * CZ;       // + 8,388,608 floats
  hipMemcpyAsync(Mo, d_in[0], (size_t)S * R * CM * 4, hipMemcpyDeviceToDevice, stream);
  hipMemcpyAsync(Zo, d_in[1], (size_t)R * R * CZ * 4, hipMemcpyDeviceToDevice, stream);
  const float* msa_mask  = f(2);
  const float* pair_mask = f(3);

  // -------- workspace layout --------
  char* WB    = (char*)d_ws;                       //   8 MB bf16 weight arena
  char* slabA = WB + ((size_t)8 << 20);            // 256 MB scores / outer / hidden
  char* slabB = slabA + (size_t)268435456;         // 128 MB probs / perm / einsum
  char* slabC = slabB + (size_t)134217728;         // per-phase cursor

  // -------- weight conversion (f32 [K,N] -> bf16 [N,K]) --------
  enum { W_MR_Q, W_MR_K, W_MR_V, W_MR_G, W_MR_O, W_MR_ZB,
         W_MT_L1, W_MT_L2,
         W_OPM_A, W_OPM_B, W_OPM_OUT,
         W_PT_L1, W_PT_L2,
         W_TAS_Q, W_TAS_K, W_TAS_V, W_TAS_G, W_TAS_O, W_TAS_TB,
         W_TAE_Q, W_TAE_K, W_TAE_V, W_TAE_G, W_TAE_O, W_TAE_TB,
         W_TMO_AP, W_TMO_AG, W_TMO_BP, W_TMO_BG, W_TMO_G, W_TMO_P,
         W_TMI_AP, W_TMI_AG, W_TMI_BP, W_TMI_BG, W_TMI_G, W_TMI_P,
         W_S, W_COUNT };
  unsigned short* wt[W_COUNT];
  size_t wboff = 0;
  auto conv = [&](int slot, int pidx, int K, int N) {
    unsigned short* dst = (unsigned short*)(WB + wboff);
    wboff += ((size_t)K * N * 2 + 255) & ~(size_t)255;
    long long total = (long long)K * N;
    k_wT<<<(unsigned)((total + 255) / 256), 256, 0, stream>>>(f(pidx), dst, K, N, total);
    wt[slot] = dst;
  };
  conv(W_MR_Q, 13, 256, 256);  conv(W_MR_K, 6, 256, 256);
  conv(W_MR_V, 14, 256, 256);  conv(W_MR_G, 5, 256, 256);
  conv(W_MR_O, 12, 256, 256);  conv(W_MR_ZB, 15, 128, 8);
  conv(W_MT_L1, 17, 256, 1024); conv(W_MT_L2, 19, 1024, 256);
  conv(W_OPM_A, 23, 256, 32);  conv(W_OPM_B, 25, 256, 32);
  conv(W_OPM_OUT, 29, 1024, 128);
  conv(W_PT_L1, 31, 128, 512); conv(W_PT_L2, 33, 512, 128);
  conv(W_TAS_Q, 53, 128, 128); conv(W_TAS_K, 48, 128, 128);
  conv(W_TAS_V, 55, 128, 128); conv(W_TAS_G, 47, 128, 128);
  conv(W_TAS_O, 52, 128, 128); conv(W_TAS_TB, 54, 128, 4);
  conv(W_TAE_Q, 43, 128, 128); conv(W_TAE_K, 38, 128, 128);
  conv(W_TAE_V, 45, 128, 128); conv(W_TAE_G, 37, 128, 128);
  conv(W_TAE_O, 42, 128, 128); conv(W_TAE_TB, 44, 128, 4);
  conv(W_TMO_AP, 75, 128, 128); conv(W_TMO_AG, 73, 128, 128);
  conv(W_TMO_BP, 79, 128, 128); conv(W_TMO_BG, 77, 128, 128);
  conv(W_TMO_G, 81, 128, 128);  conv(W_TMO_P, 87, 128, 128);
  conv(W_TMI_AP, 59, 128, 128); conv(W_TMI_AG, 57, 128, 128);
  conv(W_TMI_BP, 63, 128, 128); conv(W_TMI_BG, 61, 128, 128);
  conv(W_TMI_G, 65, 128, 128);  conv(W_TMI_P, 71, 128, 128);
  conv(W_S, 89, 256, 384);

  size_t cur = 0;
  auto al = [&](size_t bytes) { void* p = slabC + cur; cur += (bytes + 255) & ~(size_t)255; return p; };

  // ================= 1) MSA row attention (H=8, c=32) =================
  {
    cur = 0;
    unsigned short* lnm   = (unsigned short*)al(8388608);   // [16384,256] bf16
    unsigned short* zln   = (unsigned short*)al(16777216);  // [65536,128] bf16
    float*          zbias = (float*)al(2097152);            // [65536,8]
    unsigned short* qb    = (unsigned short*)al(8388608);
    unsigned short* kb    = (unsigned short*)al(8388608);
    unsigned short* vb    = (unsigned short*)al(8388608);
    float*          gate  = (float*)al(16777216);
    unsigned short* vT    = (unsigned short*)al(8388608);
    float*          att   = (float*)al(16777216);
    unsigned short* gated = (unsigned short*)al(8388608);
    float*          scores = (float*)slabA;
    unsigned short* probs  = (unsigned short*)slabB;

    k_ln<<<2048, 256, 0, stream>>>(Mo, f(8), f(7), lnm, 256);    // ln_m
    k_ln<<<8192, 256, 0, stream>>>(Zo, f(10), f(9), zln, 128);   // ln_z
    gemm(stream, zln, 128, 0, 0, wt[W_MR_ZB], 128, 0, 0, nullptr, zbias, nullptr,
         8, 0, 0, 1, 65536, 8, 128, 1, 1.f, 0, 0);
    gemm(stream, lnm, 256, 0, 0, wt[W_MR_Q], 256, 0, 0, nullptr, nullptr, qb,
         256, 0, 0, 1, 16384, 256, 256, 1, 1.f, 0, 0);
    gemm(stream, lnm, 256, 0, 0, wt[W_MR_K], 256, 0, 0, nullptr, nullptr, kb,
         256, 0, 0, 1, 16384, 256, 256, 1, 1.f, 0, 0);
    gemm(stream, lnm, 256, 0, 0, wt[W_MR_V], 256, 0, 0, nullptr, nullptr, vb,
         256, 0, 0, 1, 16384, 256, 256, 1, 1.f, 0, 0);
    gemm(stream, lnm, 256, 0, 0, wt[W_MR_G], 256, 0, 0, f(4), gate, nullptr,
         256, 0, 0, 1, 16384, 256, 256, 1, 1.f, 2, 0);
    gemm(stream, qb, 256, 65536, 32, kb, 256, 65536, 32, nullptr, scores, nullptr,
         256, 524288, 65536, 8, 256, 256, 32, 512, 0.17677669529663687f, 0, 0);
    k_softmax<<<16384, 256, 0, stream>>>(scores, probs, msa_mask, 256, 0, zbias, 8);
    k_head_pack<<<16384, 256, 0, stream>>>(vb, vT, 256, 8, 32, 4194304LL);
    gemm(stream, probs, 256, 524288, 65536, vT, 256, 65536, 8192, nullptr, att, nullptr,
         256, 65536, 32, 8, 256, 32, 256, 512, 1.f, 0, 0);
    k_mul_b<<<16384, 256, 0, stream>>>(att, gate, gated, 4194304LL);
    gemm(stream, gated, 256, 0, 0, wt[W_MR_O], 256, 0, 0, f(11), Mo, nullptr,
         256, 0, 0, 1, 16384, 256, 256, 1, 1.f, 0, 1);
  }

  // ================= 2) MSA transition =================
  {
    cur = 0;
    unsigned short* lnb = (unsigned short*)al(8388608);
    unsigned short* h1  = (unsigned short*)al(33554432);    // [16384,1024] bf16
    float*          tmp = (float*)al(16777216);
    k_ln<<<2048, 256, 0, stream>>>(Mo, f(21), f(20), lnb, 256);
    gemm(stream, lnb, 256, 0, 0, wt[W_MT_L1], 256, 0, 0, f(16), nullptr, h1,
         1024, 0, 0, 1, 16384, 1024, 256, 1, 1.f, 1, 0);
    gemm(stream, h1, 1024, 0, 0, wt[W_MT_L2], 1024, 0, 0, f(18), tmp, nullptr,
         256, 0, 0, 1, 16384, 256, 1024, 1, 1.f, 0, 0);
    k_add_masked<<<16384, 256, 0, stream>>>(Mo, tmp, msa_mask, 256, 4194304LL);
  }

  // ================= 3) Outer product mean =================
  {
    cur = 0;
    unsigned short* lnb  = (unsigned short*)al(8388608);
    float*          aF   = (float*)al(2097152);
    float*          bF   = (float*)al(2097152);
    unsigned short* aT   = (unsigned short*)al(1048576);    // [8192,64] bf16
    unsigned short* bT   = (unsigned short*)al(1048576);
    float*          opmo = (float*)al(33554432);
    float*          nrm  = (float*)al(262144);
    float*          outer = (float*)slabA;                  // [8192,8192] f32
    unsigned short* outp  = (unsigned short*)slabB;         // [65536,1024] bf16

    k_ln<<<2048, 256, 0, stream>>>(Mo, f(27), f(26), lnb, 256);
    gemm(stream, lnb, 256, 0, 0, wt[W_OPM_A], 256, 0, 0, f(22), aF, nullptr,
         32, 0, 0, 1, 16384, 32, 256, 1, 1.f, 0, 0);
    gemm(stream, lnb, 256, 0, 0, wt[W_OPM_B], 256, 0, 0, f(24), bF, nullptr,
         32, 0, 0, 1, 16384, 32, 256, 1, 1.f, 0, 0);
    k_opm_pack<<<2048, 256, 0, stream>>>(aF, msa_mask, aT, 524288LL);
    k_opm_pack<<<2048, 256, 0, stream>>>(bF, msa_mask, bT, 524288LL);
    gemm(stream, aT, 64, 0, 0, bT, 64, 0, 0, nullptr, outer, nullptr,
         8192, 0, 0, 1, 8192, 8192, 64, 1, 1.f, 0, 0);
    k_opm_perm<<<262144, 256, 0, stream>>>(outer, outp, 67108864LL);
    gemm(stream, outp, 1024, 0, 0, wt[W_OPM_OUT], 1024, 0, 0, f(28), opmo, nullptr,
         128, 0, 0, 1, 65536, 128, 1024, 1, 1.f, 0, 0);
    k_norm<<<256, 256, 0, stream>>>(msa_mask, nrm);
    k_opm_add<<<32768, 256, 0, stream>>>(Zo, opmo, nrm, 8388608LL);
  }

  // ================= 4) triangle mult outgoing (tmo, base 72) =================
  tri_mul_run(stream, Zo, pair_mask, f(82), f(83),
              wt[W_TMO_AP], f(74), wt[W_TMO_AG], f(72),
              wt[W_TMO_BP], f(78), wt[W_TMO_BG], f(76),
              wt[W_TMO_G], f(80), f(84), f(85),
              wt[W_TMO_P], f(86), 0, slabA, slabB, slabC);

  // ================= 5) triangle mult incoming (tmi, base 56) =================
  tri_mul_run(stream, Zo, pair_mask, f(66), f(67),
              wt[W_TMI_AP], f(58), wt[W_TMI_AG], f(56),
              wt[W_TMI_BP], f(62), wt[W_TMI_BG], f(60),
              wt[W_TMI_G], f(64), f(68), f(69),
              wt[W_TMI_P], f(70), 1, slabA, slabB, slabC);

  // ================= 6) triangle attention starting node (tas, base 46) ========
  tri_att_run(stream, Zo, Zo, pair_mask, 0, f(49), f(50),
              wt[W_TAS_Q], wt[W_TAS_K], wt[W_TAS_V], wt[W_TAS_G], f(46),
              wt[W_TAS_O], f(51), wt[W_TAS_TB], slabA, slabB, slabC);

  // ================= 7) triangle attention ending node (tae, base 36) ==========
  {
    float* zt = (float*)slabC;                              // 33.5 MB
    k_transpose3d<<<32768, 256, 0, stream>>>(Zo, zt, 8388608LL);
    tri_att_run(stream, zt, zt, pair_mask, 1, f(39), f(40),
                wt[W_TAE_Q], wt[W_TAE_K], wt[W_TAE_V], wt[W_TAE_G], f(36),
                wt[W_TAE_O], f(41), wt[W_TAE_TB],
                slabA, slabB, slabC + (size_t)33554432);
    k_transpose3d<<<32768, 256, 0, stream>>>(zt, Zo, 8388608LL);
  }

  // ================= 8) pair transition =================
  {
    cur = 0;
    unsigned short* lnb = (unsigned short*)al(16777216);
    float*          tmp = (float*)al(33554432);
    unsigned short* h1  = (unsigned short*)slabA;           // [65536,512] bf16
    k_ln<<<8192, 256, 0, stream>>>(Zo, f(35), f(34), lnb, 128);
    gemm(stream, lnb, 128, 0, 0, wt[W_PT_L1], 128, 0, 0, f(30), nullptr, h1,
         512, 0, 0, 1, 65536, 512, 128, 1, 1.f, 1, 0);
    gemm(stream, h1, 512, 0, 0, wt[W_PT_L2], 512, 0, 0, f(32), tmp, nullptr,
         128, 0, 0, 1, 65536, 128, 512, 1, 1.f, 0, 0);
    k_add_masked<<<32768, 256, 0, stream>>>(Zo, tmp, pair_mask, 128, 8388608LL);
  }

  // ================= 9) single embedding: s = linear(m[0]) =================
  {
    unsigned short* m0b = (unsigned short*)slabC;
    k_cvt<<<256, 256, 0, stream>>>(Mo, m0b, 65536LL);
    gemm(stream, m0b, 256, 0, 0, wt[W_S], 256, 0, 0, f(88), So, nullptr,
         384, 0, 0, 1, 256, 384, 256, 1, 1.f, 0, 0);
  }
}
